// FeatureFlipFusion_28604482191967
// MI455X (gfx1250) — matrix operand order repaired and checked by reading, not compiled, not run on hardware
//
#include <hip/hip_runtime.h>
#include <hip/hip_bf16.h>
#include <math.h>

// ---------------------------------------------------------------------------
// FeatureFlipFusion on gfx1250 (MI455X): bf16 WMMA for all three GEMMs.
// B=8, C=256, H=W=64, K=3 (KK=9).
// Round 2: wider N tiles (dcn/proj: N=32/block, off: N=64/block) to amortize
// A-fragment loads and sampling-record setup over more WMMAs per K-chunk.
// ---------------------------------------------------------------------------

typedef __attribute__((ext_vector_type(16))) __bf16 v16bf;
typedef __attribute__((ext_vector_type(8)))  float  v8f;

#define C_CH   256
#define HW     4096
#define WDIM   64
#define BSZ    8
#define KKTAPS 9
#define KDCN   (KKTAPS * C_CH)       // 2304
#define KOFF   (KKTAPS * 2 * C_CH)   // 4608
#define BN_EPS 1e-5f

// ---- WMMA fragment helpers (ISA 7.12.2 layouts, wave32) --------------------

// A-matrix 16x32 bf16, row-major source A[m][k] with row stride `ld`.
__device__ __forceinline__ v16bf load_a_frag_bf16(const __bf16* A, int ld, int K0) {
    int lane = threadIdx.x & 31;
    int m    = lane & 15;
    int half = lane >> 4;
    const __bf16* r0 = A + (size_t)m * ld + K0 + half * 8;       // K group 0
    const __bf16* r1 = r0 + 16;                                  // K group 1
    v16bf a;
#pragma unroll
    for (int i = 0; i < 8; ++i) { a[i] = r0[i]; a[8 + i] = r1[i]; }
    return a;
}

// Same A layout but sourced from fp32 global memory with on-the-fly cvt.
__device__ __forceinline__ v16bf load_a_frag_f32(const float* A, int ld, int K0) {
    int lane = threadIdx.x & 31;
    int m    = lane & 15;
    int half = lane >> 4;
    const float* r0 = A + (size_t)m * ld + K0 + half * 8;
    const float* r1 = r0 + 16;
    v16bf a;
#pragma unroll
    for (int i = 0; i < 8; ++i) { a[i] = (__bf16)r0[i]; a[8 + i] = (__bf16)r1[i]; }
    return a;
}

// B-matrix 32x16 bf16 from an LDS tile with row stride `ld`: n = lane%16,
// elements i<8 at k = 8*half+i, elements 8+i at k = 16+8*half+i.
__device__ __forceinline__ v16bf load_b_frag_lds(const __bf16* Bt, int ld) {
    int lane = threadIdx.x & 31;
    int n    = lane & 15;
    int half = lane >> 4;
    v16bf b;
#pragma unroll
    for (int i = 0; i < 8; ++i) { b[i] = Bt[(8 * half + i) * ld + n]; b[8 + i] = Bt[(16 + 8 * half + i) * ld + n]; }
    return b;
}

// B-matrix 32x16 from fp32 global: n = lane%16, element (k, n) at Bp[k*ld + n], same k order as A.
__device__ __forceinline__ v16bf load_b_frag_f32g(const float* Bp, int ld) {
    int lane = threadIdx.x & 31;
    int n    = lane & 15;
    int half = lane >> 4;
    v16bf b;
#pragma unroll
    for (int i = 0; i < 8; ++i) { b[i] = (__bf16)Bp[(size_t)(8 * half + i) * ld + n]; b[8 + i] = (__bf16)Bp[(size_t)(16 + 8 * half + i) * ld + n]; }
    return b;
}

#define WMMA_BF16(A, B, Cacc) \
    __builtin_amdgcn_wmma_f32_16x16x32_bf16(false, (A), false, (B), (short)0, (Cacc), false, false)

// ---- weight repack kernels -------------------------------------------------

// dcn_w[O=256][C=256][9] -> Apack[o][K], K = tap*256 + c (tap-major K order)
__global__ void repack_dcnw(const float* __restrict__ w, __bf16* __restrict__ ap) {
    int i = blockIdx.x * 256 + threadIdx.x;
    if (i >= C_CH * KDCN) return;
    int o = i / KDCN, K = i % KDCN;
    int tap = K / C_CH, c = K % C_CH;
    ap[i] = (__bf16)w[(size_t)(o * C_CH + c) * KKTAPS + tap];
}

// off_w[27][512][3][3] -> Apack[32][K], K = tap*512 + ci; rows 27..31 zero.
__global__ void repack_offw(const float* __restrict__ w, __bf16* __restrict__ ap) {
    int i = blockIdx.x * 256 + threadIdx.x;
    if (i >= 32 * KOFF) return;
    int m = i / KOFF, K = i % KOFF;
    float v = 0.f;
    if (m < 27) {
        int tap = K / (2 * C_CH), ci = K % (2 * C_CH);
        v = w[(size_t)(m * 2 * C_CH + ci) * KKTAPS + tap];
    }
    ap[i] = (__bf16)v;
}

// ---- proj = BN1(conv1x1(feature)) : M=256, K=256, N=32/block ---------------

__global__ __launch_bounds__(256) void proj_gemm_bn(
    const float* __restrict__ feat, const float* __restrict__ w,
    const float* __restrict__ bias,
    const float* __restrict__ g, const float* __restrict__ bb,
    const float* __restrict__ mm, const float* __restrict__ vv,
    float* __restrict__ proj)
{
    int hw0  = blockIdx.x * 32;
    int b    = blockIdx.y;
    int wave = threadIdx.x >> 5;
    int lane = threadIdx.x & 31;
    int m0   = wave * 32;                       // each wave: two 16-row M tiles

    const float* Bbase = feat + (size_t)b * C_CH * HW + hw0;
    v8f acc[2][2];
    acc[0][0] = (v8f)0.f; acc[0][1] = (v8f)0.f;
    acc[1][0] = (v8f)0.f; acc[1][1] = (v8f)0.f;

    for (int K0 = 0; K0 < C_CH; K0 += 32) {
        v16bf b0 = load_b_frag_f32g(Bbase + (size_t)K0 * HW,      HW);
        v16bf b1 = load_b_frag_f32g(Bbase + (size_t)K0 * HW + 16, HW);
        v16bf a0 = load_a_frag_f32(w + (size_t)m0 * C_CH,        C_CH, K0);
        v16bf a1 = load_a_frag_f32(w + (size_t)(m0 + 16) * C_CH, C_CH, K0);
        acc[0][0] = WMMA_BF16(a0, b0, acc[0][0]);
        acc[0][1] = WMMA_BF16(a0, b1, acc[0][1]);
        acc[1][0] = WMMA_BF16(a1, b0, acc[1][0]);
        acc[1][1] = WMMA_BF16(a1, b1, acc[1][1]);
    }

    int n = lane & 15, mh = (lane >> 4) * 8;
#pragma unroll
    for (int t = 0; t < 2; ++t) {
        int mb = m0 + t * 16 + mh;
#pragma unroll
        for (int i = 0; i < 8; ++i) {
            int co = mb + i;
            float inv = g[co] * rsqrtf(vv[co] + BN_EPS);
            float sh  = bb[co] - mm[co] * inv;
            float bi  = bias[co];
#pragma unroll
            for (int u = 0; u < 2; ++u) {
                float val = (acc[t][u][i] + bi) * inv + sh;
                proj[((size_t)b * C_CH + co) * HW + hw0 + u * 16 + n] = val;
            }
        }
    }
}

// ---- off = conv3x3(concat(flip(feature), proj)) : M=32(pad27), K=4608 ------
// One full 64-pixel row per block: 8 waves = 2 M-tiles x 4 N-tiles.
// Output channels 0..8 = dy (raw), 9..17 = dx (raw), 18..26 = sigmoid -> mask.

__global__ __launch_bounds__(256) void off_gemm(
    const float* __restrict__ feat, const float* __restrict__ proj,
    const __bf16* __restrict__ ap, const float* __restrict__ off_b,
    float* __restrict__ offall)
{
    __shared__ __bf16 Bt[32 * 64];               // [k=32][n=64]
    int h    = blockIdx.x;                       // one image row per block
    int b    = blockIdx.y;
    int hw0  = h * WDIM;
    int tid  = threadIdx.x;
    int wave = tid >> 5, lane = tid & 31;
    int m0   = (wave & 1) * 16;                  // M tile
    int nt   = wave >> 1;                        // N tile 0..3

    const float* fplane = feat + (size_t)b * C_CH * HW;
    const float* pplane = proj + (size_t)b * C_CH * HW;
    v8f acc = (v8f)0.f;

    for (int tap = 0; tap < KKTAPS; ++tap) {
        int ky = tap / 3 - 1, kx = tap % 3 - 1;
        int iy = h + ky;
        for (int c0 = 0; c0 < 2 * C_CH; c0 += 32) {
            __syncthreads();                     // protect previous tile reads
#pragma unroll
            for (int r = 0; r < 8; ++r) {        // 2048 vals / 256 threads
                int e = tid + 256 * r;
                int n = e & 63, kk = e >> 6;
                int c = c0 + kk;
                int ix = n + kx;
                float v = 0.f;
                if (iy >= 0 && iy < WDIM && ix >= 0 && ix < WDIM) {
                    v = (c < C_CH)
                      ? fplane[(size_t)c * HW + iy * WDIM + (WDIM - 1 - ix)]  // flip
                      : pplane[(size_t)(c - C_CH) * HW + iy * WDIM + ix];
                }
                Bt[kk * 64 + n] = (__bf16)v;
            }
            __syncthreads();
            int K0 = tap * 2 * C_CH + c0;
            v16bf bf = load_b_frag_lds(Bt + nt * 16, 64);
            v16bf a  = load_a_frag_bf16(ap + (size_t)m0 * KOFF, KOFF, K0);
            acc = WMMA_BF16(a, bf, acc);
        }
    }

    int n = lane & 15, mh = (lane >> 4) * 8;
#pragma unroll
    for (int i = 0; i < 8; ++i) {
        int ch = m0 + mh + i;
        if (ch < 27) {
            float v = acc[i] + off_b[ch];
            if (ch >= 18) v = 1.f / (1.f + __expf(-v));   // mask channels
            offall[((size_t)b * 27 + ch) * HW + hw0 + nt * 16 + n] = v;
        }
    }
}

// ---- deformable sample + einsum + BN2 + residual ReLU ----------------------
// Per block: 32 pixels, M=256, K=2304 (tap-major). Bilinear coords (with the
// W-flip folded in, corner weights pre-scaled by validity and mask) cached
// in LDS; each K-chunk builds a 32x32 bf16 B-tile via 4-corner gathers, then
// each wave runs 2 M-tiles x 2 N-tiles = 4 WMMAs.

__global__ __launch_bounds__(256) void dcn_gemm(
    const float* __restrict__ feat, const float* __restrict__ proj,
    const float* __restrict__ offall, const __bf16* __restrict__ ap,
    const float* __restrict__ dcn_b,
    const float* __restrict__ g, const float* __restrict__ bb,
    const float* __restrict__ mm, const float* __restrict__ vv,
    float* __restrict__ out)
{
    __shared__ int    soff[288][4];              // 32 px x 9 taps
    __shared__ float  swt [288][4];
    __shared__ __bf16 Bt[32 * 32];               // [k=32][n=32]

    int hw0 = blockIdx.x * 32;
    int b   = blockIdx.y;
    int h   = hw0 >> 6, w0 = hw0 & 63;
    int tid = threadIdx.x;
    const float* fplane = feat + (size_t)b * C_CH * HW;

    for (int e = tid; e < 288; e += 256) {       // sampling records
        int n = e / 9, tap = e % 9;
        const float* ob = offall + (size_t)b * 27 * HW + hw0 + n;
        float dyv = ob[(size_t)tap * HW];
        float dxv = ob[(size_t)(9 + tap) * HW];
        float mk  = ob[(size_t)(18 + tap) * HW];
        float py = (float)h + (float)(tap / 3 - 1) + dyv;
        float px = (float)(w0 + n) + (float)(tap % 3 - 1) + dxv;
        float y0f = floorf(py), x0f = floorf(px);
        float wy = py - y0f, wx = px - x0f;
        int y0 = (int)y0f, x0 = (int)x0f;
        int   ys[2] = { y0, y0 + 1 }, xs[2] = { x0, x0 + 1 };
        float wyv[2] = { 1.f - wy, wy }, wxv[2] = { 1.f - wx, wx };
#pragma unroll
        for (int cy = 0; cy < 2; ++cy)
#pragma unroll
            for (int cx = 0; cx < 2; ++cx) {
                int idx = cy * 2 + cx;
                int yy = ys[cy], xx = xs[cx];
                bool valid = (yy >= 0 && yy < WDIM && xx >= 0 && xx < WDIM);
                soff[e][idx] = valid ? (yy * WDIM + (WDIM - 1 - xx)) : 0;    // flip
                swt [e][idx] = valid ? (wyv[cy] * wxv[cx] * mk) : 0.f;
            }
    }
    __syncthreads();

    int wave = tid >> 5, lane = tid & 31;
    int m0 = wave * 32;                          // 8 waves x 2 M-tiles = M 256
    v8f acc[2][2];
    acc[0][0] = (v8f)0.f; acc[0][1] = (v8f)0.f;
    acc[1][0] = (v8f)0.f; acc[1][1] = (v8f)0.f;

    for (int tap = 0; tap < KKTAPS; ++tap) {
        for (int c0 = 0; c0 < C_CH; c0 += 32) {
            __syncthreads();
#pragma unroll
            for (int r = 0; r < 4; ++r) {        // 1024 vals / 256 threads
                int e = tid + 256 * r;
                int n = e & 31, kk = e >> 5;
                int c = c0 + kk;
                int si = n * 9 + tap;
                const float* cp = fplane + (size_t)c * HW;
                float v = swt[si][0] * cp[soff[si][0]]
                        + swt[si][1] * cp[soff[si][1]]
                        + swt[si][2] * cp[soff[si][2]]
                        + swt[si][3] * cp[soff[si][3]];
                Bt[kk * 32 + n] = (__bf16)v;
            }
            __syncthreads();
            int K0 = tap * C_CH + c0;
            v16bf b0 = load_b_frag_lds(Bt,      32);
            v16bf b1 = load_b_frag_lds(Bt + 16, 32);
            v16bf a0 = load_a_frag_bf16(ap + (size_t)m0 * KDCN,        KDCN, K0);
            v16bf a1 = load_a_frag_bf16(ap + (size_t)(m0 + 16) * KDCN, KDCN, K0);
            acc[0][0] = WMMA_BF16(a0, b0, acc[0][0]);
            acc[0][1] = WMMA_BF16(a0, b1, acc[0][1]);
            acc[1][0] = WMMA_BF16(a1, b0, acc[1][0]);
            acc[1][1] = WMMA_BF16(a1, b1, acc[1][1]);
        }
    }

    int n = lane & 15, mh = (lane >> 4) * 8;
#pragma unroll
    for (int t = 0; t < 2; ++t) {
        int mb = m0 + t * 16 + mh;
#pragma unroll
        for (int i = 0; i < 8; ++i) {
            int co = mb + i;
            float inv = g[co] * rsqrtf(vv[co] + BN_EPS);
            float sh  = bb[co] - mm[co] * inv;
            float bi  = dcn_b[co];
#pragma unroll
            for (int u = 0; u < 2; ++u) {
                float y = (acc[t][u][i] + bi) * inv + sh;
                size_t o = ((size_t)b * C_CH + co) * HW + hw0 + u * 16 + n;
                float r = proj[o] + y;
                out[o] = r > 0.f ? r : 0.f;
            }
        }
    }
}

// ---------------------------------------------------------------------------

extern "C" void kernel_launch(void* const* d_in, const int* in_sizes, int n_in,
                              void* d_out, int out_size, void* d_ws, size_t ws_size,
                              hipStream_t stream) {
    const float* feature = (const float*)d_in[0];
    const float* p1_w    = (const float*)d_in[1];
    const float* p1_b    = (const float*)d_in[2];
    const float* bn1_g   = (const float*)d_in[3];
    const float* bn1_b   = (const float*)d_in[4];
    const float* bn1_m   = (const float*)d_in[5];
    const float* bn1_v   = (const float*)d_in[6];
    const float* off_w   = (const float*)d_in[7];
    const float* off_b   = (const float*)d_in[8];
    const float* dcn_w   = (const float*)d_in[9];
    const float* dcn_b   = (const float*)d_in[10];
    const float* bn2_g   = (const float*)d_in[11];
    const float* bn2_b   = (const float*)d_in[12];
    const float* bn2_m   = (const float*)d_in[13];
    const float* bn2_v   = (const float*)d_in[14];
    float* out = (float*)d_out;

    char* ws = (char*)d_ws;
    size_t cur = 0;
    auto take = [&](size_t bytes) { size_t p = cur; cur += (bytes + 255) & ~(size_t)255; return p; };
    float*  proj    = (float*) (ws + take((size_t)BSZ * C_CH * HW * sizeof(float)));  // 33.5 MB
    float*  offall  = (float*) (ws + take((size_t)BSZ * 27  * HW * sizeof(float)));   //  3.5 MB
    __bf16* ap_dcn  = (__bf16*)(ws + take((size_t)C_CH * KDCN * sizeof(__bf16)));     //  1.2 MB
    __bf16* ap_off  = (__bf16*)(ws + take((size_t)32   * KOFF * sizeof(__bf16)));     //  0.3 MB

    repack_dcnw<<<(C_CH * KDCN + 255) / 256, 256, 0, stream>>>(dcn_w, ap_dcn);
    repack_offw<<<(32 * KOFF + 255) / 256, 256, 0, stream>>>(off_w, ap_off);

    proj_gemm_bn<<<dim3(HW / 32, BSZ), 256, 0, stream>>>(feature, p1_w, p1_b,
                                                         bn1_g, bn1_b, bn1_m, bn1_v, proj);
    off_gemm<<<dim3(WDIM, BSZ), 256, 0, stream>>>(feature, proj, ap_off, off_b, offall);
    dcn_gemm<<<dim3(HW / 32, BSZ), 256, 0, stream>>>(feature, proj, offall, ap_dcn, dcn_b,
                                                     bn2_g, bn2_b, bn2_m, bn2_v, out);
}
